// GaussianLayer_12884901888050
// MI455X (gfx1250) — matrix operand order repaired and checked by reading, not compiled, not run on hardware
//
#include <hip/hip_runtime.h>
#include <hip/hip_bf16.h>
#include <math.h>

#define IN_F   4096
#define OUT_F  4096
#define NCOMP  256
#define TWO_PI_F 6.28318530717958647692f

typedef __attribute__((ext_vector_type(16))) _Float16 v16h;
typedef __attribute__((ext_vector_type(8)))  _Float16 v8h;
typedef __attribute__((ext_vector_type(8)))  float    v8f;
typedef __attribute__((ext_vector_type(4)))  float    v4f;

// ---------------------------------------------------------------------------
// Kernel 0: inv_norm = 1 / (IN_F * sum(weights))
// ---------------------------------------------------------------------------
__global__ __launch_bounds__(256) void sumw_kernel(const float* __restrict__ w,
                                                   float* __restrict__ inv_norm) {
    __shared__ float s[256];
    int t = threadIdx.x;
    s[t] = w[t];
    __syncthreads();
    for (int off = 128; off > 0; off >>= 1) {
        if (t < off) s[t] += s[t + off];
        __syncthreads();
    }
    if (t == 0) inv_norm[0] = 1.0f / ((float)IN_F * s[0]);
}

// ---------------------------------------------------------------------------
// Kernel 1: G_inT[m][i] = exp(-(xi-mu0)^2/(2 v0)) / sqrt(2 pi v0) * w[m]  (f16)
// Stored transposed (M rows, I cols) so GEMM1 B-fragments are contiguous in K.
// ---------------------------------------------------------------------------
__global__ __launch_bounds__(256) void gin_kernel(const float* __restrict__ mus,
                                                  const float* __restrict__ log_vars,
                                                  const float* __restrict__ weights,
                                                  _Float16* __restrict__ GinT) {
    int idx = blockIdx.x * 256 + threadIdx.x;      // [0, NCOMP*IN_F)
    int mc  = idx >> 12;                           // component
    int i   = idx & (IN_F - 1);
    float xi  = (float)i * (1.0f / (float)(IN_F - 1));
    float mu  = mus[2 * mc];
    float var = expf(log_vars[2 * mc]);
    float d   = xi - mu;
    float g   = expf(-d * d / (2.0f * var)) * rsqrtf(TWO_PI_F * var) * weights[mc];
    GinT[(size_t)mc * IN_F + i] = (_Float16)g;
}

// ---------------------------------------------------------------------------
// Kernel 2: G_outT[o][m] = pdf(x_o; mu1, v1) * inv_norm   (f16, transposed)
// ---------------------------------------------------------------------------
__global__ __launch_bounds__(256) void gout_kernel(const float* __restrict__ mus,
                                                   const float* __restrict__ log_vars,
                                                   const float* __restrict__ inv_norm,
                                                   _Float16* __restrict__ GoutT) {
    int idx = blockIdx.x * 256 + threadIdx.x;      // [0, OUT_F*NCOMP)
    int o   = idx >> 8;
    int mc  = idx & (NCOMP - 1);
    float xo  = (float)o * (1.0f / (float)(OUT_F - 1));
    float mu  = mus[2 * mc + 1];
    float var = expf(log_vars[2 * mc + 1]);
    float d   = xo - mu;
    float g   = expf(-d * d / (2.0f * var)) * rsqrtf(TWO_PI_F * var) * inv_norm[0];
    GoutT[(size_t)o * NCOMP + mc] = (_Float16)g;
}

// ---------------------------------------------------------------------------
// WMMA fragment helpers (wave32; A/B in contiguous K order, C/D per ISA 7.12.2)
// A 16x32 f16: lane row = lane&15; half = lane>>4
//   elems 0..15 -> K = half*16 + (0..15)
// B 32x16 f16: lane col = lane&15; elems 0..15 -> K = half*16 + (0..15)
// C/D 16x16 f32: elem j -> row = j + half*8, col = lane&15
// ---------------------------------------------------------------------------
__device__ __forceinline__ v16h load_a_frag_f32(const float* p) {
    // p points at (row, k0 + half*16); consecutive f32, converted to f16
    const v4f* q = (const v4f*)p;
    v4f a0 = q[0], a1 = q[1];          // K: half*16 + 0..7
    v4f a2 = q[2], a3 = q[3];          // K: half*16 + 8..15  (p+8)
    v16h a;
#pragma unroll
    for (int i = 0; i < 4; ++i) {
        a[i]      = (_Float16)a0[i];
        a[4 + i]  = (_Float16)a1[i];
        a[8 + i]  = (_Float16)a2[i];
        a[12 + i] = (_Float16)a3[i];
    }
    return a;
}

__device__ __forceinline__ v16h load_a_frag_f16(const _Float16* p) {
    // p points at (row, k0 + half*16); 16 contiguous f16
    v8h lo = *(const v8h*)(p);
    v8h hi = *(const v8h*)(p + 8);
    v16h a;
#pragma unroll
    for (int i = 0; i < 8; ++i) { a[i] = lo[i]; a[8 + i] = hi[i]; }
    return a;
}

// ---------------------------------------------------------------------------
// GEMM1: C[B,256] = f16( f32 X[B,4096] @ G_in[4096,256] )  via WMMA f16->f32
// Block: 256 threads = 8 waves; wave tile 16x128; block tile 64x256.
// All 8 B-fragments are loaded before the WMMA burst so load latency
// overlaps matrix execution within the wave (staged s_wait_loadcnt).
// ---------------------------------------------------------------------------
__global__ __launch_bounds__(256) void gemm1_kernel(const float* __restrict__ X,
                                                    const _Float16* __restrict__ GinT,
                                                    _Float16* __restrict__ C) {
    const int lane = threadIdx.x & 31;
    const int wave = threadIdx.x >> 5;
    const int wy = wave >> 1;          // 0..3
    const int wx = wave & 1;           // 0..1
    const int lr = lane & 15;
    const int hl = lane >> 4;
    const int r0 = blockIdx.x * 64 + wy * 16;
    const int n0 = wx * 128;

    v8f acc[8];
#pragma unroll
    for (int t = 0; t < 8; ++t) acc[t] = (v8f)0.0f;

    const float*    xrow = X + (size_t)(r0 + lr) * IN_F + hl * 16;
    const _Float16* brow = GinT + (size_t)(n0 + lr) * IN_F + hl * 16;

    for (int k0 = 0; k0 < IN_F; k0 += 32) {
        // Issue all B loads up front (one clause), then A, then WMMA burst.
        v16h b[8];
#pragma unroll
        for (int t = 0; t < 8; ++t)
            b[t] = *(const v16h*)(brow + (size_t)t * 16 * IN_F + k0);
        v16h a = load_a_frag_f32(xrow + k0);
#pragma unroll
        for (int t = 0; t < 8; ++t)
            acc[t] = __builtin_amdgcn_wmma_f32_16x16x32_f16(
                false, a, false, b[t], (short)0, acc[t], false, false);
    }

#pragma unroll
    for (int t = 0; t < 8; ++t) {
        int col = n0 + t * 16 + lr;
#pragma unroll
        for (int j = 0; j < 8; ++j) {
            int row = r0 + j + hl * 8;
            C[(size_t)row * NCOMP + col] = (_Float16)acc[t][j];
        }
    }
}

// ---------------------------------------------------------------------------
// GEMM2: out[B,4096] = f32( C[B,256] @ G_out[256,4096] )  via WMMA f16->f32
// Wave tile 16x128; block tile 64x256; K=256 fully unrolled.
// ---------------------------------------------------------------------------
__global__ __launch_bounds__(256) void gemm2_kernel(const _Float16* __restrict__ C,
                                                    const _Float16* __restrict__ GoutT,
                                                    float* __restrict__ out) {
    const int lane = threadIdx.x & 31;
    const int wave = threadIdx.x >> 5;
    const int wy = wave >> 1;
    const int wx = wave & 1;
    const int lr = lane & 15;
    const int hl = lane >> 4;
    const int r0 = blockIdx.x * 64 + wy * 16;
    const int c0 = blockIdx.y * 256 + wx * 128;

    v8f acc[8];
#pragma unroll
    for (int t = 0; t < 8; ++t) acc[t] = (v8f)0.0f;

    const _Float16* crow = C + (size_t)(r0 + lr) * NCOMP + hl * 16;
    const _Float16* brow = GoutT + (size_t)(c0 + lr) * NCOMP + hl * 16;

#pragma unroll
    for (int k0 = 0; k0 < NCOMP; k0 += 32) {
        v16h b[8];
#pragma unroll
        for (int t = 0; t < 8; ++t)
            b[t] = *(const v16h*)(brow + (size_t)t * 16 * NCOMP + k0);
        v16h a = load_a_frag_f16(crow + k0);
#pragma unroll
        for (int t = 0; t < 8; ++t)
            acc[t] = __builtin_amdgcn_wmma_f32_16x16x32_f16(
                false, a, false, b[t], (short)0, acc[t], false, false);
    }

#pragma unroll
    for (int t = 0; t < 8; ++t) {
        int col = c0 + t * 16 + lr;
#pragma unroll
        for (int j = 0; j < 8; ++j) {
            int row = r0 + j + hl * 8;
            out[(size_t)row * OUT_F + col] = acc[t][j];
        }
    }
}

// ---------------------------------------------------------------------------
extern "C" void kernel_launch(void* const* d_in, const int* in_sizes, int n_in,
                              void* d_out, int out_size, void* d_ws, size_t ws_size,
                              hipStream_t stream) {
    const float* x        = (const float*)d_in[0];
    const float* mus      = (const float*)d_in[1];
    const float* log_vars = (const float*)d_in[2];
    const float* weights  = (const float*)d_in[3];

    const int rows = in_sizes[0] / IN_F;   // batch (4096)

    char* ws = (char*)d_ws;
    float*    inv_norm = (float*)ws;                                    // 256 B slot
    _Float16* GinT  = (_Float16*)(ws + 256);                            // 2 MB
    _Float16* GoutT = (_Float16*)(ws + 256 + (size_t)NCOMP * IN_F * 2); // 2 MB
    _Float16* Cbuf  = (_Float16*)(ws + 256 + (size_t)NCOMP * IN_F * 2
                                          + (size_t)OUT_F * NCOMP * 2); // rows*256*2

    sumw_kernel<<<1, 256, 0, stream>>>(weights, inv_norm);
    gin_kernel <<<(NCOMP * IN_F) / 256, 256, 0, stream>>>(mus, log_vars, weights, GinT);
    gout_kernel<<<(OUT_F * NCOMP) / 256, 256, 0, stream>>>(mus, log_vars, inv_norm, GoutT);
    gemm1_kernel<<<rows / 64, 256, 0, stream>>>(x, GinT, Cbuf);
    gemm2_kernel<<<dim3(rows / 64, OUT_F / 256), 256, 0, stream>>>(Cbuf, GoutT,
                                                                   (float*)d_out);
}
